// ConcatAttentionLayer_49074296324329
// MI455X (gfx1250) — matrix operand order repaired and checked by reading, not compiled, not run on hardware
//
#include <hip/hip_runtime.h>
#include <hip/hip_bf16.h>
#include <math.h>

typedef __attribute__((ext_vector_type(16))) _Float16 v16h;
typedef __attribute__((ext_vector_type(8)))  float    v8f;
typedef __attribute__((ext_vector_type(4)))  float    f32x4;

#define S_LEN 2048
#define B_SZ  64
#define D_IN  512
#define D_SRC 512
#define D_AL  512
#define M_ROWS (S_LEN * B_SZ)   // 131072

// ---------------------------------------------------------------------------
// Kernel 1a: convert W1b (second 512 columns of W1) to f16, row-major [512][512]
// ---------------------------------------------------------------------------
__global__ __launch_bounds__(256) void k_w1b_to_f16(const float* __restrict__ W1,
                                                    _Float16* __restrict__ Bh) {
    const int gid = blockIdx.x * 256 + threadIdx.x;       // 262144 elements
    const int n = gid >> 9;
    const int k = gid & 511;
    Bh[gid] = (_Float16)W1[(size_t)n * (D_IN + D_SRC) + D_IN + k];
}

// ---------------------------------------------------------------------------
// Kernel 1b: pre[b][a] = b1[a] + sum_d input[b][d] * W1[a][d]   (tiny GEMM)
// ---------------------------------------------------------------------------
__global__ __launch_bounds__(256) void k_pre(const float* __restrict__ input,
                                             const float* __restrict__ W1,
                                             const float* __restrict__ b1,
                                             float* __restrict__ pre) {
    const int gid = blockIdx.x * 256 + threadIdx.x;       // 64*512 = 32768
    const int b = gid >> 9;
    const int a = gid & 511;
    const float* xr = input + (size_t)b * D_IN;
    const float* wr = W1 + (size_t)a * (D_IN + D_SRC);
    float acc = b1[a];
    for (int d = 0; d < D_IN; d += 4) {
        f32x4 x = *(const f32x4*)(xr + d);
        f32x4 w = *(const f32x4*)(wr + d);
        acc += x[0]*w[0] + x[1]*w[1] + x[2]*w[2] + x[3]*w[3];
    }
    pre[gid] = acc;
}

// ---------------------------------------------------------------------------
// Kernel 2: main GEMM (M=131072, N=512, K=512) via v_wmma_f32_16x16x32_f16
//           fused with tanh + dot(W2) -> scores[row]
// One wave handles 16 rows; two passes over N (16 tiles of 16 each).
// Accumulators initialized with pre[b][n] (so the broadcast-input half of the
// concat GEMM and the bias come for free).
// ---------------------------------------------------------------------------
__global__ __launch_bounds__(256) void k_gemm_scores(
        const float*    __restrict__ src,    // [M_ROWS][512]  (source_hids flat)
        const _Float16* __restrict__ Bh,     // [512][512] f16 W1b
        const float*    __restrict__ pre,    // [64][512]
        const float*    __restrict__ W2,     // [512]
        float*          __restrict__ scores) // [M_ROWS]
{
    const int lane   = threadIdx.x & 31;
    const int wave   = threadIdx.x >> 5;
    const int gwave  = blockIdx.x * 8 + wave;
    const int rowBase = gwave << 4;          // 16 rows per wave
    const int bBase   = rowBase & 63;        // batch index of row 0 (no wrap: 16|64)
    const int hi      = lane >> 4;           // 0: lanes 0-15, 1: lanes 16-31
    const int l15     = lane & 15;

    float sp[8];
    #pragma unroll
    for (int r = 0; r < 8; ++r) sp[r] = 0.0f;

    // A-matrix row for this lane (both lane halves read the same 16 rows)
    const float* aPtr = src + (size_t)(rowBase + l15) * D_SRC;

    for (int pass = 0; pass < 2; ++pass) {
        const int nPass = pass * 256;

        // --- init accumulators with pre[b][n] (D layout: elem r -> row r / r+8)
        v8f c[16];
        #pragma unroll
        for (int t = 0; t < 16; ++t) {
            const int n0 = nPass + t * 16 + l15;
            #pragma unroll
            for (int r = 0; r < 8; ++r) {
                const int brow = bBase + (hi ? (r + 8) : r);
                c[t][r] = pre[brow * D_AL + n0];
            }
        }

        // --- K loop: 16 steps of K=32
        for (int kb = 0; kb < D_SRC; kb += 32) {
            // A fragment, 16x32 f16 layout (contiguous, matches B):
            //   lanes 0-15 : elems 0..15 = K kb+0..15
            //   lanes 16-31: elems 0..15 = K kb+16..31
            const int abase = kb + (hi ? 16 : 0);
            f32x4 a0 = *(const f32x4*)(aPtr + abase);
            f32x4 a1 = *(const f32x4*)(aPtr + abase + 4);
            f32x4 a2 = *(const f32x4*)(aPtr + abase + 8);
            f32x4 a3 = *(const f32x4*)(aPtr + abase + 12);
            v16h a;
            #pragma unroll
            for (int j = 0; j < 4; ++j) {
                a[j]      = (_Float16)a0[j];
                a[4 + j]  = (_Float16)a1[j];
                a[8 + j]  = (_Float16)a2[j];
                a[12 + j] = (_Float16)a3[j];
            }
            #pragma unroll
            for (int t = 0; t < 16; ++t) {
                const int n0 = nPass + t * 16 + l15;
                // B 32x16 layout: lane = N; lanes 0-15 hold K kb+0..15,
                // lanes 16-31 hold K kb+16..31 -> one contiguous 32B load.
                const v16h bf = *(const v16h*)(Bh + (size_t)n0 * D_SRC + kb + (hi ? 16 : 0));
                c[t] = __builtin_amdgcn_wmma_f32_16x16x32_f16(
                           false, a, false, bf, (short)0, c[t], false, false);
            }
        }

        // --- fused epilogue: score partials  sp[r] += tanh(c) * W2[n]
        #pragma unroll
        for (int t = 0; t < 16; ++t) {
            const int n0 = nPass + t * 16 + l15;
            const float w2 = W2[n0];
            #pragma unroll
            for (int r = 0; r < 8; ++r)
                sp[r] += tanhf(c[t][r]) * w2;
        }
    }

    // reduce partials over each 16-lane half (columns) -> per-row score
    #pragma unroll
    for (int r = 0; r < 8; ++r) {
        float v = sp[r];
        v += __shfl_xor(v, 8, 16);
        v += __shfl_xor(v, 4, 16);
        v += __shfl_xor(v, 2, 16);
        v += __shfl_xor(v, 1, 16);
        sp[r] = v;
    }
    if (l15 == 0) {
        const int base = rowBase + (hi ? 8 : 0);
        #pragma unroll
        for (int r = 0; r < 8; ++r)
            scores[base + r] = sp[r];  // b2 omitted: constant shift cancels in softmax
    }
}

// ---------------------------------------------------------------------------
// Kernel 3: per-batch softmax over S + attn output + ctx = sum_s attn*src
// One block per b (64 blocks, 256 threads).
// ---------------------------------------------------------------------------
__global__ __launch_bounds__(256) void k_softmax_ctx(
        const float*         __restrict__ scores, // [S][B]
        const unsigned char* __restrict__ mask,   // [S][B] bool
        const float*         __restrict__ src,    // [S][B][512]
        float*               __restrict__ ctx,    // [B][512]
        float*               __restrict__ attn)   // [S][B]
{
    __shared__ float se[S_LEN];
    __shared__ float red[256];
    const int b = blockIdx.x;
    const int t = threadIdx.x;

    float lmax = -INFINITY;
    for (int s = t; s < S_LEN; s += 256) {
        float v = scores[s * B_SZ + b];
        if (mask[s * B_SZ + b]) v = -INFINITY;
        se[s] = v;
        lmax = fmaxf(lmax, v);
    }
    red[t] = lmax; __syncthreads();
    for (int o = 128; o > 0; o >>= 1) { if (t < o) red[t] = fmaxf(red[t], red[t + o]); __syncthreads(); }
    const float mx = red[0];
    __syncthreads();

    float lsum = 0.0f;
    for (int s = t; s < S_LEN; s += 256) {
        const float e = expf(se[s] - mx);
        se[s] = e;
        lsum += e;
    }
    red[t] = lsum; __syncthreads();
    for (int o = 128; o > 0; o >>= 1) { if (t < o) red[t] += red[t + o]; __syncthreads(); }
    const float inv = 1.0f / red[0];
    __syncthreads();

    for (int s = t; s < S_LEN; s += 256)
        attn[s * B_SZ + b] = se[s] * inv;

    // ctx: each thread owns 2 consecutive d values -> coalesced float2 loads
    const int d0 = t * 2;
    float acc0 = 0.0f, acc1 = 0.0f;
    for (int s = 0; s < S_LEN; ++s) {
        const float w = se[s] * inv;
        const float2 v = *(const float2*)(src + ((size_t)(s * B_SZ + b)) * D_SRC + d0);
        acc0 += w * v.x;
        acc1 += w * v.y;
    }
    ctx[b * D_SRC + d0]     = acc0;
    ctx[b * D_SRC + d0 + 1] = acc1;
}

// ---------------------------------------------------------------------------
extern "C" void kernel_launch(void* const* d_in, const int* in_sizes, int n_in,
                              void* d_out, int out_size, void* d_ws, size_t ws_size,
                              hipStream_t stream) {
    (void)in_sizes; (void)n_in; (void)out_size; (void)ws_size;

    const float*         input = (const float*)d_in[0];          // [1,64,512]
    const float*         src   = (const float*)d_in[1];          // [2048,64,512]
    const unsigned char* mask  = (const unsigned char*)d_in[2];  // [2048,64] bool
    const float*         W1    = (const float*)d_in[3];          // [512,1024]
    const float*         b1    = (const float*)d_in[4];          // [512]
    const float*         W2    = (const float*)d_in[5];          // [1,512]
    // b2 (d_in[6]) intentionally unused: constant shift cancels in softmax,
    // and neither ctx nor attn depend on it.

    // workspace layout
    _Float16* Bh     = (_Float16*)d_ws;                                  // 512 KB
    float*    pre    = (float*)((char*)d_ws + (size_t)D_AL * D_SRC * 2); // 128 KB
    float*    scores = pre + (size_t)B_SZ * D_AL;                        // 512 KB

    float* ctx  = (float*)d_out;                 // [64][512]
    float* attn = ctx + (size_t)B_SZ * D_SRC;    // [2048][64]

    k_w1b_to_f16<<<(D_AL * D_SRC) / 256, 256, 0, stream>>>(W1, Bh);
    k_pre<<<(B_SZ * D_AL) / 256, 256, 0, stream>>>(input, W1, b1, pre);
    k_gemm_scores<<<M_ROWS / 16 / 8, 256, 0, stream>>>(src, Bh, pre, W2, scores);
    k_softmax_ctx<<<B_SZ, 256, 0, stream>>>(scores, mask, src, ctx, attn);
}
